// GENConvModel_21534966022658
// MI455X (gfx1250) — matrix operand order repaired and checked by reading, not compiled, not run on hardware
//
#include <hip/hip_runtime.h>
#include <hip/hip_bf16.h>

// ---------------------------------------------------------------------------
// GENConv GNN model on gfx1250 (MI455X).
//   - All dense GEMMs use v_wmma_f32_16x16x32_bf16 (wave32, 16x16 C tiles,
//     4 N-tiles per wave for A-fragment reuse). K must be a multiple of 32;
//     the two ragged-K operands (edge_attr K=16, mol_features K=200) are
//     zero-padded into workspace so the hot loop has no guards/divergence.
//   - Edge softmax-aggregation = 3 bandwidth-bound atomic passes over the
//     300Kx256 message tensor (the true bottleneck at 23.3 TB/s).
//   - Weights converted once per launch to bf16, transposed [N, Kpad] so
//     B fragments are contiguous per-lane loads.
// Workspace requirement: ~730 MB (message tensor dominates).
// ---------------------------------------------------------------------------

#define NNODES 50000
#define NEDGES 300000
#define NGRAPH 2048
#define DC     256
#define EXPC   512
#define CATC   384
#define EPSM   1e-7f
#define BNEPS  1e-5f

typedef __attribute__((ext_vector_type(16))) __bf16 v16bf;
typedef __attribute__((ext_vector_type(8)))  float  v8f;

// ---------------------------------------------------------------------------
// WMMA GEMM: C[M,N] = A[M,K] (f32, row-major) @ B[K,N]
// B supplied pre-transposed+padded as Bt[N, Kpad] bf16 (zeros for k>=K).
// mode 0: plain store
// mode 1: C = relu(C + bias[n])
// mode 2: C = relu(C + gsrc[gidx[row]*gld + n]) + EPSM      (GENConv message)
// Requirements: M % 16 == 0, N % 64 == 0, K % 32 == 0 (no tail path; EXEC
// stays all-ones as WMMA requires — the only branch is wave-uniform).
// ---------------------------------------------------------------------------
__global__ void __launch_bounds__(256)
k_wmma_gemm(const float* __restrict__ A, int lda,
            const __bf16* __restrict__ Bt, int ldb /* = Kpad */,
            float* __restrict__ C, int ldc,
            const float* __restrict__ bias,
            const int* __restrict__ gidx, const float* __restrict__ gsrc, int gld,
            int M, int N, int K, int mode)
{
    const int lane  = threadIdx.x & 31;
    const int wave  = threadIdx.x >> 5;          // wave-uniform
    const int strip = blockIdx.x * 8 + wave;
    if (strip * 16 >= M) return;                 // wave-uniform cull: EXEC stays full
    const int row0 = strip * 16;
    const int col0 = blockIdx.y * 64;

    const int mrow = lane & 15;                  // A: M = lane&15
    const int kh   = lane >> 4;                  // A/B: K-half select
    const int ncol = lane & 15;                  // B/C: N = lane&15

    v8f acc0 = {}, acc1 = {}, acc2 = {}, acc3 = {};
    const float* arow = A + (long)(row0 + mrow) * lda;

    for (int k0 = 0; k0 < K; k0 += 32) {
        // ---- A fragment: 16x32 bf16, ISA 7.12.2 layout ----
        // lanes 0-15: K = k0+{0..7} and k0+16+{0..7}
        // lanes 16-31: K = k0+8+{0..7} and k0+24+{0..7}
        const float* ap0 = arow + k0 + kh * 8;        // 8 consecutive f32 (b128 x2)
        const float* ap1 = arow + k0 + 16 + kh * 8;
        v16bf a;
#pragma unroll
        for (int e = 0; e < 8; ++e) a[e]     = (__bf16)ap0[e];
#pragma unroll
        for (int e = 0; e < 8; ++e) a[8 + e] = (__bf16)ap1[e];

        __builtin_prefetch(arow + k0 + 32, 0, 3);   // global_prefetch_b8, keep near

        // ---- 4 B fragments / 4 WMMAs (A reused) ----
#pragma unroll
        for (int t = 0; t < 4; ++t) {
            const __bf16* bp = Bt + (long)(col0 + t * 16 + ncol) * ldb + k0 + kh * 8;
            v16bf b;
#pragma unroll
            for (int e = 0; e < 8; ++e) { b[e] = bp[e]; b[8 + e] = bp[16 + e]; }  // 2 x 16 contiguous bytes
            v8f* accp = (t == 0) ? &acc0 : (t == 1) ? &acc1 : (t == 2) ? &acc2 : &acc3;
            *accp = __builtin_amdgcn_wmma_f32_16x16x32_bf16(
                        false, a, false, b, (short)0, *accp, false, false);
        }
    }

    // ---- epilogue: C layout = lanes give N, VGPR v gives M = (lane>>4)*8 + v ----
#pragma unroll
    for (int t = 0; t < 4; ++t) {
        v8f acc = (t == 0) ? acc0 : (t == 1) ? acc1 : (t == 2) ? acc2 : acc3;
        const int cc = col0 + t * 16 + ncol;
        const float bv = bias ? bias[cc] : 0.0f;
#pragma unroll
        for (int v = 0; v < 8; ++v) {
            const int cr = row0 + kh * 8 + v;
            float val = acc[v];
            if (mode == 1) {
                val += bv;
                val = val > 0.0f ? val : 0.0f;
            } else if (mode == 2) {
                val += gsrc[(long)gidx[cr] * gld + cc];
                val = (val > 0.0f ? val : 0.0f) + EPSM;
            }
            C[(long)cr * ldc + cc] = val;
        }
    }
}

// Convert f32 weight W[K,N] -> bf16 transposed Bt[N,Kpad] (zero pad k>=K)
__global__ void k_convt(const float* __restrict__ W, __bf16* __restrict__ Bt,
                        int K, int N, int Kpad)
{
    long i = (long)blockIdx.x * blockDim.x + threadIdx.x;
    long tot = (long)N * Kpad;
    if (i >= tot) return;
    int n = (int)(i / Kpad);
    int k = (int)(i % Kpad);
    Bt[i] = (k < K) ? (__bf16)W[(long)k * N + n] : (__bf16)0.0f;
}

// Zero-padded row copy: dst[r, 0:Kdst] = {src[r, 0:Ksrc], 0...}
__global__ void k_pad(const float* __restrict__ src, float* __restrict__ dst,
                      int Ksrc, int Kdst, long total)
{
    long i = (long)blockIdx.x * blockDim.x + threadIdx.x;
    if (i >= total) return;
    int r = (int)(i / Kdst);
    int k = (int)(i % Kdst);
    dst[i] = (k < Ksrc) ? src[(long)r * Ksrc + k] : 0.0f;
}

__global__ void k_fill(float* __restrict__ p, long n, float v)
{
    long i = (long)blockIdx.x * blockDim.x + threadIdx.x;
    if (i < n) p[i] = v;
}

// segment max over destinations; messages are strictly positive so integer
// atomicMax on the float bit pattern (init 0) is exact.
__global__ void k_seg_max(const float* __restrict__ m, const int* __restrict__ dst,
                          int* __restrict__ mx, long total)
{
    long i = (long)blockIdx.x * blockDim.x + threadIdx.x;
    if (i >= total) return;
    int e = (int)(i >> 8), c = (int)(i & 255);
    atomicMax(&mx[(long)dst[e] * DC + c], __float_as_int(m[i]));
}

__global__ void k_seg_sum(const float* __restrict__ m, const int* __restrict__ dst,
                          const int* __restrict__ mx,
                          float* __restrict__ den, float* __restrict__ num, long total)
{
    long i = (long)blockIdx.x * blockDim.x + threadIdx.x;
    if (i >= total) return;
    int e = (int)(i >> 8), c = (int)(i & 255);
    long o = (long)dst[e] * DC + c;
    float mv = m[i];
    float ex = __expf(mv - __int_as_float(mx[o]));
    atomicAdd(&den[o], ex);
    atomicAdd(&num[o], mv * ex);
}

__global__ void k_agg(const float* __restrict__ den, const float* __restrict__ num,
                      const float* __restrict__ hin, float* __restrict__ z, long total)
{
    long i = (long)blockIdx.x * blockDim.x + threadIdx.x;
    if (i >= total) return;
    float d = den[i];
    z[i] = num[i] / (d > 0.0f ? d : 1.0f) + hin[i];
}

// per-column sum / sumsq over T[M,512] -> stats[0:512]=sum, [512:1024]=sumsq
__global__ void k_colstats(const float* __restrict__ T, int M, float* __restrict__ stats)
{
    int c0 = threadIdx.x;                       // columns c0 and c0+256
    long r0 = (long)blockIdx.x * 64;
    float s0 = 0.f, s1 = 0.f, q0 = 0.f, q1 = 0.f;
    for (int r = 0; r < 64; ++r) {
        long row = r0 + r;
        if (row >= M) break;
        float v0 = T[row * EXPC + c0];
        float v1 = T[row * EXPC + c0 + 256];
        s0 += v0; q0 += v0 * v0;
        s1 += v1; q1 += v1 * v1;
    }
    atomicAdd(&stats[c0],              s0);
    atomicAdd(&stats[c0 + 256],        s1);
    atomicAdd(&stats[EXPC + c0],       q0);
    atomicAdd(&stats[EXPC + c0 + 256], q1);
}

__global__ void k_bn_relu(float* __restrict__ T, const float* __restrict__ stats,
                          const float* __restrict__ g, const float* __restrict__ b,
                          int M, long total)
{
    long i = (long)blockIdx.x * blockDim.x + threadIdx.x;
    if (i >= total) return;
    int c = (int)(i & (EXPC - 1));
    float mu  = stats[c] / (float)M;
    float var = stats[EXPC + c] / (float)M - mu * mu;
    float v = (T[i] - mu) * rsqrtf(var + BNEPS) * g[c] + b[c];
    T[i] = v > 0.0f ? v : 0.0f;
}

__global__ void k_pool_sum(const float* __restrict__ h, const int* __restrict__ batch,
                           float* __restrict__ cat, float* __restrict__ cnt, long total)
{
    long i = (long)blockIdx.x * blockDim.x + threadIdx.x;
    if (i >= total) return;
    int node = (int)(i >> 8), c = (int)(i & 255);
    int gidx = batch[node];
    atomicAdd(&cat[(long)gidx * CATC + c], h[i]);
    if (c == 0) atomicAdd(&cnt[gidx], 1.0f);
}

__global__ void k_pool_div(float* __restrict__ cat, const float* __restrict__ cnt, long total)
{
    long i = (long)blockIdx.x * blockDim.x + threadIdx.x;
    if (i >= total) return;
    int gidx = (int)(i >> 8), c = (int)(i & 255);
    float n = cnt[gidx];
    cat[(long)gidx * CATC + c] /= (n > 0.0f ? n : 1.0f);
}

__global__ void k_head(const float* __restrict__ H, const float* __restrict__ Wo,
                       const float* __restrict__ bo, float* __restrict__ out, int G)
{
    int gi = blockIdx.x * blockDim.x + threadIdx.x;
    if (gi >= G) return;
    const float* h = H + (long)gi * 512;
    float s = 0.0f;
    for (int k = 0; k < 512; ++k) s += h[k] * Wo[k];
    out[gi] = s + bo[0];
}

// ---------------------------------------------------------------------------
extern "C" void kernel_launch(void* const* d_in, const int* in_sizes, int n_in,
                              void* d_out, int out_size, void* d_ws, size_t ws_size,
                              hipStream_t stream)
{
    (void)in_sizes; (void)n_in; (void)out_size; (void)ws_size;

    const float* x    = (const float*)d_in[0];
    const float* eatt = (const float*)d_in[1];
    const float* mol  = (const float*)d_in[2];
    const int*   eidx = (const int*)d_in[36];
    const int*   src  = eidx;
    const int*   dst  = eidx + NEDGES;
    const int*   bat  = (const int*)d_in[37];

    // ---- workspace carve-out ----
    size_t off = 0;
    auto take = [&](size_t bytes) -> void* {
        void* p = (char*)d_ws + off;
        off += (bytes + 255) & ~(size_t)255;
        return p;
    };
    float* mbuf  = (float*)take((size_t)NEDGES * DC * 4);   // 307 MB messages
    float* mxf   = (float*)take((size_t)NNODES * DC * 4);   // seg max (int-punned)
    float* den   = (float*)take((size_t)NNODES * DC * 4);
    float* num   = (float*)take((size_t)NNODES * DC * 4);
    float* n0    = (float*)take((size_t)NNODES * DC * 4);   // z / h'
    float* n1    = (float*)take((size_t)NNODES * DC * 4);   // layer output
    float* tbuf  = (float*)take((size_t)NNODES * EXPC * 4); // 102 MB
    float* stats = (float*)take(1024 * 4);
    float* cat   = (float*)take((size_t)NGRAPH * CATC * 4);
    float* h2a   = (float*)take((size_t)NGRAPH * 512 * 4);
    float* h2b   = (float*)take((size_t)NGRAPH * 512 * 4);
    float* cnt   = (float*)take((size_t)NGRAPH * 4);
    float* epad  = (float*)take((size_t)NEDGES * 32 * 4);   // edge_attr padded K=32
    float* mpad  = (float*)take((size_t)NGRAPH * 224 * 4);  // mol_features padded K=224

    __bf16* WetT[3];  __bf16* W1tT[3];  __bf16* W2tT[3];  __bf16* WltT[3];
    for (int l = 0; l < 3; ++l) {
        WetT[l] = (__bf16*)take((size_t)256 * 32 * 2);
        W1tT[l] = (__bf16*)take((size_t)512 * 256 * 2);
        W2tT[l] = (__bf16*)take((size_t)256 * 512 * 2);
        WltT[l] = (__bf16*)take((size_t)256 * 256 * 2);
    }
    __bf16* mlp1t = (__bf16*)take((size_t)512 * 224 * 2);
    __bf16* mlp2t = (__bf16*)take((size_t)512 * 512 * 2);
    __bf16* mlp3t = (__bf16*)take((size_t)128 * 512 * 2);
    __bf16* pr1t  = (__bf16*)take((size_t)512 * 384 * 2);
    __bf16* pr2t  = (__bf16*)take((size_t)512 * 512 * 2);

    auto fill = [&](float* p, long n, float v) {
        k_fill<<<dim3((unsigned)((n + 255) / 256)), 256, 0, stream>>>(p, n, v);
    };
    auto convt = [&](const float* W, __bf16* Bt, int K, int N) {
        int Kpad = (K + 31) & ~31;
        long tot = (long)N * Kpad;
        k_convt<<<dim3((unsigned)((tot + 255) / 256)), 256, 0, stream>>>(W, Bt, K, N, Kpad);
    };
    auto gemm = [&](const float* A, int lda, const __bf16* Bt, int Kpad,
                    float* C, int ldc, const float* bias,
                    const int* gidx, const float* gsrc, int gld,
                    int M, int N, int K, int mode) {
        dim3 grid((unsigned)((M / 16 + 7) / 8), (unsigned)(N / 64));
        k_wmma_gemm<<<grid, 256, 0, stream>>>(A, lda, Bt, Kpad, C, ldc, bias,
                                              gidx, gsrc, gld, M, N, K, mode);
    };

    // ---- weight conversion (bf16, transposed, K padded to 32) ----
    for (int l = 0; l < 3; ++l) {
        convt((const float*)d_in[3 + 5 * l + 0], WetT[l], 16,  256);
        convt((const float*)d_in[3 + 5 * l + 1], W1tT[l], 256, 512);
        convt((const float*)d_in[3 + 5 * l + 2], W2tT[l], 512, 256);
        convt((const float*)d_in[18 + 2 * l],    WltT[l], 256, 256);
    }
    convt((const float*)d_in[24], mlp1t, 200, 512);
    convt((const float*)d_in[26], mlp2t, 512, 512);
    convt((const float*)d_in[28], mlp3t, 512, 128);
    convt((const float*)d_in[30], pr1t,  384, 512);
    convt((const float*)d_in[32], pr2t,  512, 512);

    // ---- pad ragged-K A operands (removes all guards from the WMMA loop) ----
    {
        long tot = (long)NEDGES * 32;
        k_pad<<<dim3((unsigned)((tot + 255) / 256)), 256, 0, stream>>>(eatt, epad, 16, 32, tot);
        long tot2 = (long)NGRAPH * 224;
        k_pad<<<dim3((unsigned)((tot2 + 255) / 256)), 256, 0, stream>>>(mol, mpad, 200, 224, tot2);
    }

    const long edgeTot = (long)NEDGES * DC;   // 76.8M
    const long nodeTot = (long)NNODES * DC;   // 12.8M
    const unsigned eBlocks = (unsigned)((edgeTot + 255) / 256);
    const unsigned nBlocks = (unsigned)((nodeTot + 255) / 256);

    // ---- 3x (GENConv + Linear) ----
    const float* hin = x;
    for (int l = 0; l < 3; ++l) {
        const float* bng  = (const float*)d_in[3 + 5 * l + 3];
        const float* bnb  = (const float*)d_in[3 + 5 * l + 4];
        const float* linb = (const float*)d_in[19 + 2 * l];

        // m = relu(edge_attr @ We + h[src]) + eps       [E, 256]
        gemm(epad, 32, WetT[l], 32, mbuf, DC, nullptr, src, hin, DC,
             NEDGES, DC, 32, 2);

        // softmax aggregation over incoming edges
        fill(mxf, nodeTot, 0.0f);
        k_seg_max<<<eBlocks, 256, 0, stream>>>(mbuf, dst, (int*)mxf, edgeTot);
        fill(den, nodeTot, 0.0f);
        fill(num, nodeTot, 0.0f);
        k_seg_sum<<<eBlocks, 256, 0, stream>>>(mbuf, dst, (const int*)mxf, den, num, edgeTot);

        // z = agg + h  (residual)
        k_agg<<<nBlocks, 256, 0, stream>>>(den, num, hin, n0, nodeTot);

        // t = z @ W1 ; batchnorm(batch stats) ; relu
        gemm(n0, DC, W1tT[l], 256, tbuf, EXPC, nullptr, nullptr, nullptr, 0,
             NNODES, EXPC, DC, 0);
        fill(stats, 1024, 0.0f);
        k_colstats<<<dim3((NNODES + 63) / 64), 256, 0, stream>>>(tbuf, NNODES, stats);
        k_bn_relu<<<dim3((unsigned)(((long)NNODES * EXPC + 255) / 256)), 256, 0, stream>>>(
            tbuf, stats, bng, bnb, NNODES, (long)NNODES * EXPC);

        // h' = t @ W2
        gemm(tbuf, EXPC, W2tT[l], 512, n0, DC, nullptr, nullptr, nullptr, 0,
             NNODES, DC, EXPC, 0);

        // h = relu(h' @ Wl + b)
        gemm(n0, DC, WltT[l], 256, n1, DC, linb, nullptr, nullptr, 0,
             NNODES, DC, DC, 1);
        hin = n1;
    }

    // ---- global mean pool into concat[:, 0:256] ----
    fill(cat, (long)NGRAPH * CATC, 0.0f);
    fill(cnt, NGRAPH, 0.0f);
    k_pool_sum<<<nBlocks, 256, 0, stream>>>(hin, bat, cat, cnt, nodeTot);
    k_pool_div<<<dim3((NGRAPH * DC + 255) / 256), 256, 0, stream>>>(cat, cnt, (long)NGRAPH * DC);

    // ---- MLP branch -> concat[:, 256:384] ----
    gemm(mpad, 224, mlp1t, 224, h2a, 512, (const float*)d_in[25], nullptr, nullptr, 0,
         NGRAPH, 512, 224, 1);
    gemm(h2a, 512, mlp2t, 512, h2b, 512, (const float*)d_in[27], nullptr, nullptr, 0,
         NGRAPH, 512, 512, 1);
    gemm(h2b, 512, mlp3t, 512, cat + 256, CATC, (const float*)d_in[29], nullptr, nullptr, 0,
         NGRAPH, 128, 512, 1);

    // ---- predictor ----
    gemm(cat, CATC, pr1t, 384, h2a, 512, (const float*)d_in[31], nullptr, nullptr, 0,
         NGRAPH, 512, 384, 1);
    gemm(h2a, 512, pr2t, 512, h2b, 512, (const float*)d_in[33], nullptr, nullptr, 0,
         NGRAPH, 512, 512, 1);

    k_head<<<dim3((NGRAPH + 255) / 256), 256, 0, stream>>>(
        h2b, (const float*)d_in[34], (const float*)d_in[35], (float*)d_out, NGRAPH);
}
